// InputPreprocessor_73040213836017
// MI455X (gfx1250) — matrix operand order repaired and checked by reading, not compiled, not run on hardware
//
#include <hip/hip_runtime.h>
#include <hip/hip_bf16.h>

typedef __bf16 v16bf __attribute__((ext_vector_type(16)));
typedef float  v8f   __attribute__((ext_vector_type(8)));

#define B_DIM 2048
#define N_EL  32
#define M_ION 8
#define D_PAIR 37
#define D_PAD  64      // K of layer 1 padded to 2 k-steps of 32
#define H_DIM  128
#define F_EL   424     // 128 + 8*37
#define W0FRAG_HALFS (2*8*32*16)   // 8192
#define W1FRAG_HALFS (4*8*32*16)   // 16384
#define WFRAG_TOTAL  (W0FRAG_HALFS + W1FRAG_HALFS)  // 24576 halfs = 48 KB

union FragBF { v16bf v; uint4 u[2]; };

__device__ __forceinline__ float silu_f(float x) {
    return x / (1.0f + __expf(-x));
}

// ---------------------------------------------------------------------------
// Prep: rearrange W0 (37x128, zero-padded K->64) and W1 (128x128) fp32 weights
// into bf16 WMMA B-fragment order: [ks][nt][lane][e], where within a k-step
// k_local = 16*(lane>>4) + e and n = nt*16 + (lane&15).   (contiguous k order, same as the A loads)
// ---------------------------------------------------------------------------
__global__ void wprep_kernel(const float* __restrict__ W0,
                             const float* __restrict__ W1,
                             __bf16* __restrict__ wfrag) {
    int idx = blockIdx.x * 256 + threadIdx.x;
    if (idx >= WFRAG_TOTAL) return;
    if (idx < W0FRAG_HALFS) {
        int e    = idx & 15;
        int lane = (idx >> 4) & 31;
        int f    = idx >> 9;          // ks*8 + nt, 0..15
        int ks = f >> 3, nt = f & 7;
        int k = ks * 32 + ((lane >> 4) << 4) + e;
        int n = nt * 16 + (lane & 15);
        float v = (k < D_PAIR) ? W0[k * H_DIM + n] : 0.0f;
        wfrag[idx] = (__bf16)v;
    } else {
        int j    = idx - W0FRAG_HALFS;
        int e    = j & 15;
        int lane = (j >> 4) & 31;
        int f    = j >> 9;            // ks*8 + nt, 0..31
        int ks = f >> 3, nt = f & 7;
        int k = ks * 32 + ((lane >> 4) << 4) + e;
        int n = nt * 16 + (lane & 15);
        wfrag[idx] = (__bf16)W1[k * H_DIM + n];
    }
}

// ---------------------------------------------------------------------------
// features_ion: silu(Z[m]*Wion[f] + bion[f]) broadcast over B.  [2048,8,32]
// ---------------------------------------------------------------------------
__global__ void ion_kernel(const float* __restrict__ Z,
                           const float* __restrict__ Wion,
                           const float* __restrict__ bion,
                           float* __restrict__ out_ion) {
    int t = threadIdx.x;          // 256 = 8 ions * 32 features
    int m = t >> 5, f = t & 31;
    float x = Z[m] * Wion[f] + bion[f];
    out_ion[(size_t)blockIdx.x * 256 + t] = silu_f(x);
}

// ---------------------------------------------------------------------------
// Main: per wave, one 16-edge tile (2 electrons x 8 ions) through the 2-layer
// MLP with v_wmma_f32_16x16x32_bf16, SiLU in fp32, ion-sum via per-lane
// accumulator reduction, plus raw-feature writeback.
// ---------------------------------------------------------------------------
__global__ __launch_bounds__(256)
void edge_mlp_kernel(const float* __restrict__ r,
                     const float* __restrict__ Rion,
                     const float* __restrict__ b0,
                     const float* __restrict__ b1,
                     const __bf16* __restrict__ wfrag,
                     float* __restrict__ out_el) {
    __shared__ __align__(16) __bf16 s_w[WFRAG_TOTAL];        // 48 KB B-fragments
    __shared__ __align__(16) __bf16 s_feat[8][16][D_PAD];    // 16 KB per-wave A stage (L1)
    __shared__ __align__(16) __bf16 s_h[8][16][H_DIM];       // 32 KB per-wave A stage (L2)

    const int t = threadIdx.x;

    // --- stage weight fragments into LDS (48 KB = 3072 uint4, 12 per thread) ---
    {
        const uint4* src = (const uint4*)wfrag;
        uint4* dst = (uint4*)s_w;
        #pragma unroll
        for (int i = 0; i < 12; ++i) dst[t + i * 256] = src[t + i * 256];
    }
    __syncthreads();

    const int lane = t & 31;
    const int w    = t >> 5;
    const int hl   = lane >> 4;     // lane-half: 0 or 1
    const int ln   = lane & 15;
    const int bn0  = blockIdx.x * 16 + w * 2;   // this wave: rows bn0, bn0+1

    // --- per-edge features: lanes 0..15 each own one edge (2 bn x 8 ions) ---
    if (lane < 16) {
        const int bn = bn0 + (lane >> 3);
        const int m  = lane & 7;
        float dx = r[bn * 3 + 0] - Rion[m * 3 + 0];
        float dy = r[bn * 3 + 1] - Rion[m * 3 + 1];
        float dz = r[bn * 3 + 2] - Rion[m * 3 + 2];
        float d2 = dx * dx + dy * dy + dz * dz;
        float d  = sqrtf(d2);
        float* op = out_el + (size_t)bn * F_EL + 128 + m * D_PAIR;
        __bf16* fp = &s_feat[w][lane][0];
        #pragma unroll
        for (int j = 0; j < 32; ++j) {
            float q      = (float)j * (1.0f / 31.0f);
            float mu     = q * q * 5.0f;
            float inv_sg = 7.0f / (1.0f + 5.0f * q);
            float tmp    = (d - mu) * inv_sg;
            float f      = d2 * __expf(-d - tmp * tmp);
            fp[j] = (__bf16)f;  op[j] = f;
        }
        float fpw = 1.0f / (d + 0.01f);
        fp[32] = (__bf16)d;   op[32] = d;
        fp[33] = (__bf16)fpw; op[33] = fpw;
        fp[34] = (__bf16)dx;  op[34] = dx;
        fp[35] = (__bf16)dy;  op[35] = dy;
        fp[36] = (__bf16)dz;  op[36] = dz;
        #pragma unroll
        for (int j = D_PAIR; j < D_PAD; ++j) fp[j] = (__bf16)0.0f;
    }
    __syncthreads();

    // --- layer 1: [16 x 64] @ [64 x 128], K = 2 steps of 32 ---
    // A layout (16-bit, 16x32): m = ln; k = ks*32 + 16*hl + e
    FragBF a1[2];
    {
        const __bf16* fr = &s_feat[w][ln][0];
        #pragma unroll
        for (int ks = 0; ks < 2; ++ks) {
            a1[ks].u[0] = *(const uint4*)(fr + ks * 32 + 16 * hl);
            a1[ks].u[1] = *(const uint4*)(fr + ks * 32 + 16 * hl + 8);
        }
    }
    v8f acc[8];
    #pragma unroll
    for (int nt = 0; nt < 8; ++nt) {
        float bv = b0[nt * 16 + ln];
        acc[nt] = (v8f){bv, bv, bv, bv, bv, bv, bv, bv};
    }
    #pragma unroll
    for (int ks = 0; ks < 2; ++ks) {
        #pragma unroll
        for (int nt = 0; nt < 8; ++nt) {
            FragBF bf;
            const __bf16* bp = s_w + ((ks * 8 + nt) * 32 + lane) * 16;
            bf.u[0] = *(const uint4*)(bp);
            bf.u[1] = *(const uint4*)(bp + 8);
            acc[nt] = __builtin_amdgcn_wmma_f32_16x16x32_bf16(
                false, a1[ks].v, false, bf.v, (short)0, acc[nt], false, false);
        }
    }

    // --- SiLU, restage h (C layout -> row-major [edge][k] for L2 A gather) ---
    #pragma unroll
    for (int nt = 0; nt < 8; ++nt) {
        #pragma unroll
        for (int i = 0; i < 8; ++i) {
            float hv = silu_f(acc[nt][i]);
            s_h[w][i + 8 * hl][nt * 16 + ln] = (__bf16)hv;   // row m = i+8*hl, col n
        }
    }
    __syncthreads();

    // --- layer 2: [16 x 128] @ [128 x 128], K = 4 steps of 32 ---
    FragBF a2[4];
    {
        const __bf16* hr = &s_h[w][ln][0];
        #pragma unroll
        for (int ks = 0; ks < 4; ++ks) {
            a2[ks].u[0] = *(const uint4*)(hr + ks * 32 + 16 * hl);
            a2[ks].u[1] = *(const uint4*)(hr + ks * 32 + 16 * hl + 8);
        }
    }
    v8f acc2[8];
    #pragma unroll
    for (int nt = 0; nt < 8; ++nt) {
        float bv = b1[nt * 16 + ln];
        acc2[nt] = (v8f){bv, bv, bv, bv, bv, bv, bv, bv};
    }
    #pragma unroll
    for (int ks = 0; ks < 4; ++ks) {
        #pragma unroll
        for (int nt = 0; nt < 8; ++nt) {
            FragBF bf;
            const __bf16* bp = s_w + W0FRAG_HALFS + ((ks * 8 + nt) * 32 + lane) * 16;
            bf.u[0] = *(const uint4*)(bp);
            bf.u[1] = *(const uint4*)(bp + 8);
            acc2[nt] = __builtin_amdgcn_wmma_f32_16x16x32_bf16(
                false, a2[ks].v, false, bf.v, (short)0, acc2[nt], false, false);
        }
    }

    // --- SiLU + sum over ions: rows 0..7 = bn0's ions, 8..15 = bn1's ions,
    //     which are exactly the 8 accumulator VGPRs of each lane-half ---
    {
        const int bn = bn0 + hl;
        float* ob = out_el + (size_t)bn * F_EL;
        #pragma unroll
        for (int nt = 0; nt < 8; ++nt) {
            float s = 0.0f;
            #pragma unroll
            for (int i = 0; i < 8; ++i) s += silu_f(acc2[nt][i]);
            ob[nt * 16 + ln] = s;   // one_el[bn][nt*16+ln]
        }
    }
}

extern "C" void kernel_launch(void* const* d_in, const int* in_sizes, int n_in,
                              void* d_out, int out_size, void* d_ws, size_t ws_size,
                              hipStream_t stream) {
    const float* r    = (const float*)d_in[0];
    const float* R    = (const float*)d_in[1];
    const float* Z    = (const float*)d_in[2];
    const float* W0   = (const float*)d_in[3];
    const float* b0   = (const float*)d_in[4];
    const float* W1   = (const float*)d_in[5];
    const float* b1   = (const float*)d_in[6];
    const float* Wion = (const float*)d_in[7];
    const float* bion = (const float*)d_in[8];

    float* out_el  = (float*)d_out;
    float* out_ion = out_el + (size_t)B_DIM * N_EL * F_EL;
    __bf16* wfrag  = (__bf16*)d_ws;

    // 1) rearrange weights into WMMA B-fragment order (bf16) in scratch
    wprep_kernel<<<(WFRAG_TOTAL + 255) / 256, 256, 0, stream>>>(W0, W1, wfrag);
    // 2) broadcast ion features
    ion_kernel<<<B_DIM, 256, 0, stream>>>(Z, Wion, bion, out_ion);
    // 3) edge features + 2-layer bf16-WMMA MLP + ion-sum + concat writeback
    edge_mlp_kernel<<<(B_DIM * N_EL) / 16, 256, 0, stream>>>(
        r, R, b0, b1, wfrag, out_el);
}
